// LSTMAttention_9174050144557
// MI455X (gfx1250) — matrix operand order repaired and checked by reading, not compiled, not run on hardware
//
#include <hip/hip_runtime.h>
#include <cstdint>
#include <cstddef>

// Problem sizes (fixed by the reference)
#define TT 2048
#define SS 2048
#define BB 16
#define DD 1024

// Block-GEMM tiling
#define BM 128          // block tile M (2 waves)
#define BN 256          // block tile N (4 waves)
#define KC 32           // K step staged in LDS
#define LDK 40          // padded LDS row stride (elements): bank-conflict-free

typedef unsigned short u16;
typedef __attribute__((ext_vector_type(16))) __bf16 bf16x16;
typedef __attribute__((ext_vector_type(8)))  float  f32x8;

// ---------------- bf16 helpers (storage treated as u16) ----------------
__device__ __forceinline__ u16 f2bf(float f) {
  union { float f; uint32_t u; } v; v.f = f;
  uint32_t r = v.u + 0x7FFFu + ((v.u >> 16) & 1u);   // round-to-nearest-even
  return (u16)(r >> 16);
}
__device__ __forceinline__ float bf2f(u16 h) {
  union { uint32_t u; float f; } v; v.u = ((uint32_t)h) << 16;
  return v.f;
}

union AFrag { uint4 u[2]; bf16x16 v; };

__device__ __forceinline__ void zero_acc(f32x8 acc[4][4]) {
  f32x8 z = {0.f, 0.f, 0.f, 0.f, 0.f, 0.f, 0.f, 0.f};
#pragma unroll
  for (int i = 0; i < 4; ++i)
#pragma unroll
    for (int j = 0; j < 4; ++j) acc[i][j] = z;
}

// -------- global -> regs prefetch of one KC-slice of the block tiles --------
// A: BM x KC (2 uint4/thread), Bt: BN x KC (4 uint4/thread); 256 threads.
__device__ __forceinline__ void stage_load_regs(const u16* __restrict__ A, int lda,
                                                const u16* __restrict__ Bt, int ldb,
                                                int kc, int tid, uint4 ra[2], uint4 rb[4]) {
  const int row = tid >> 2;            // 0..63
  const int col = (tid & 3) * 8;       // 0..24
#pragma unroll
  for (int r = 0; r < 2; ++r)
    ra[r] = *(const uint4*)(A + (size_t)(row + 64 * r) * lda + kc + col);
#pragma unroll
  for (int r = 0; r < 4; ++r)
    rb[r] = *(const uint4*)(Bt + (size_t)(row + 64 * r) * ldb + kc + col);
}

__device__ __forceinline__ void stage_store_lds(u16* __restrict__ As, u16* __restrict__ Bs,
                                                int tid, const uint4 ra[2], const uint4 rb[4]) {
  const int row = tid >> 2;
  const int col = (tid & 3) * 8;
#pragma unroll
  for (int r = 0; r < 2; ++r)
    *(uint4*)(As + (size_t)(row + 64 * r) * LDK + col) = ra[r];
#pragma unroll
  for (int r = 0; r < 4; ++r)
    *(uint4*)(Bs + (size_t)(row + 64 * r) * LDK + col) = rb[r];
}

// -------- one KC step of 64x64 wave-tile WMMA from LDS ----------------------
__device__ __forceinline__ void compute_step(const u16* __restrict__ As,
                                             const u16* __restrict__ Bs,
                                             int wm, int wn, int lane, f32x8 acc[4][4]) {
  const int r  = lane & 15;
  const int hi = lane >> 4;
  const u16* a0 = As + (size_t)(wm * 64 + r) * LDK + 8 * hi;   // A frag: K chunks 8hi, 16+8hi
  const u16* b0 = Bs + (size_t)(wn * 64 + r) * LDK + 8 * hi;   // B frag: K chunks 8hi, 16+8hi
  AFrag a[4], b[4];
#pragma unroll
  for (int i = 0; i < 4; ++i) {
    const u16* pa = a0 + (size_t)(i * 16) * LDK;
    a[i].u[0] = *(const uint4*)pa;
    a[i].u[1] = *(const uint4*)(pa + 16);
    const u16* pb = b0 + (size_t)(i * 16) * LDK;
    b[i].u[0] = *(const uint4*)pb;
    b[i].u[1] = *(const uint4*)(pb + 16);
  }
#pragma unroll
  for (int i = 0; i < 4; ++i)
#pragma unroll
    for (int j = 0; j < 4; ++j)
      acc[i][j] = __builtin_amdgcn_wmma_f32_16x16x32_bf16(
          false, a[i].v, false, b[j].v, (short)0, acc[i][j], false, false);
}

// -------- full K loop for one 128x256 block: double-buffered LDS pipeline ---
__device__ __forceinline__ void block_gemm(const u16* __restrict__ A, int lda,
                                           const u16* __restrict__ Bt, int ldb, int K,
                                           u16 (*As)[BM * LDK], u16 (*Bs)[BN * LDK],
                                           f32x8 acc[4][4]) {
  const int tid  = threadIdx.x;
  const int lane = tid & 31;
  const int wid  = tid >> 5;
  const int wm = wid >> 2, wn = wid & 3;

  uint4 ra[2], rb[4];
  stage_load_regs(A, lda, Bt, ldb, 0, tid, ra, rb);
  stage_store_lds(As[0], Bs[0], tid, ra, rb);
  __syncthreads();

  const int nsteps = K / KC;
  for (int s = 0; s < nsteps; ++s) {
    const int cur = s & 1;
    if (s + 1 < nsteps)
      stage_load_regs(A, lda, Bt, ldb, (s + 1) * KC, tid, ra, rb);
    compute_step(As[cur], Bs[cur], wm, wn, lane, acc);
    if (s + 1 < nsteps)
      stage_store_lds(As[cur ^ 1], Bs[cur ^ 1], tid, ra, rb);
    __syncthreads();
  }
}

// ---------------- prep kernels: fp32 -> bf16 with layout swaps -------------
// in [TT][BB][DD] f32 -> out [BB][TT][DD] bf16   (also used for context->ch)
__global__ void k_cvt_swap(const float* __restrict__ in, u16* __restrict__ out) {
  size_t id = (size_t)blockIdx.x * blockDim.x + threadIdx.x;
  const size_t perRow = DD / 8;
  size_t rowId = id / perRow;                 // = t*BB + b
  if (rowId >= (size_t)TT * BB) return;
  int d0 = (int)(id % perRow) * 8;
  int t = (int)(rowId / BB), b = (int)(rowId % BB);
  const float4* src = (const float4*)(in + rowId * DD + d0);
  float4 x = src[0], y = src[1];
  union { u16 h[8]; uint4 q; } o;
  o.h[0] = f2bf(x.x); o.h[1] = f2bf(x.y); o.h[2] = f2bf(x.z); o.h[3] = f2bf(x.w);
  o.h[4] = f2bf(y.x); o.h[5] = f2bf(y.y); o.h[6] = f2bf(y.z); o.h[7] = f2bf(y.w);
  *(uint4*)(out + ((size_t)b * TT + t) * DD + d0) = o.q;
}

// context [SS][BB][DD] f32 -> cT [BB][DD][SS] bf16 (full transpose for AV B-frags)
__global__ void k_cvt_T(const float* __restrict__ in, u16* __restrict__ out) {
  size_t id = (size_t)blockIdx.x * blockDim.x + threadIdx.x;
  const size_t perSeg = SS / 8;
  int s0 = (int)(id % perSeg) * 8;
  size_t tmp = id / perSeg;
  int d = (int)(tmp % DD);
  int b = (int)(tmp / DD);
  if (b >= BB) return;
  union { u16 h[8]; uint4 q; } o;
#pragma unroll
  for (int j = 0; j < 8; ++j)
    o.h[j] = f2bf(in[((size_t)(s0 + j) * BB + b) * DD + d]);
  *(uint4*)(out + ((size_t)b * DD + d) * SS + s0) = o.q;
}

// W [DD][2*DD] f32 -> bf16 same layout
__global__ void k_cvt_w(const float* __restrict__ in, u16* __restrict__ out) {
  size_t id = (size_t)blockIdx.x * blockDim.x + threadIdx.x;
  size_t n = (size_t)DD * 2 * DD / 8;
  if (id >= n) return;
  const float4* src = (const float4*)(in + id * 8);
  float4 x = src[0], y = src[1];
  union { u16 h[8]; uint4 q; } o;
  o.h[0] = f2bf(x.x); o.h[1] = f2bf(x.y); o.h[2] = f2bf(x.z); o.h[3] = f2bf(x.w);
  o.h[4] = f2bf(y.x); o.h[5] = f2bf(y.y); o.h[6] = f2bf(y.z); o.h[7] = f2bf(y.w);
  *(uint4*)(out + id * 8) = o.q;
}

// ---------------- GEMM 1: scores = Q . C^T  (per batch, bf16 out) ----------
__global__ __launch_bounds__(256) void k_gemm_qk(const u16* __restrict__ qh,
                                                 const u16* __restrict__ ch,
                                                 u16* __restrict__ sc) {
  __shared__ u16 As[2][BM * LDK];
  __shared__ u16 Bs[2][BN * LDK];
  const int NBm = TT / BM, NBn = SS / BN;   // 16, 8
  int bid = blockIdx.x;
  int b = bid / (NBm * NBn), rem = bid % (NBm * NBn);
  int mb = rem / NBn, nb = rem % NBn;

  f32x8 acc[4][4];
  zero_acc(acc);
  block_gemm(qh + (size_t)b * TT * DD + (size_t)(mb * BM) * DD, DD,
             ch + (size_t)b * SS * DD + (size_t)(nb * BN) * DD, DD,
             DD, As, Bs, acc);

  const int lane = threadIdx.x & 31, wid = threadIdx.x >> 5;
  const int wm = wid >> 2, wn = wid & 3;
  const int m0 = mb * BM + wm * 64, n0 = nb * BN + wn * 64;
  u16* C = sc + (size_t)b * TT * SS;
  const int cn = lane & 15, hi = lane >> 4;
#pragma unroll
  for (int i = 0; i < 4; ++i)
#pragma unroll
    for (int j = 0; j < 4; ++j)
#pragma unroll
      for (int rr = 0; rr < 8; ++rr)
        C[(size_t)(m0 + i * 16 + rr + 8 * hi) * SS + (n0 + j * 16 + cn)] =
            f2bf(acc[i][j][rr]);
}

// ---------------- softmax with ==0 -> -inf mask, in place (bf16) -----------
__global__ __launch_bounds__(256) void k_softmax(u16* __restrict__ P) {
  const int lane = threadIdx.x & 31;
  const int wid = threadIdx.x >> 5;
  size_t row = (size_t)blockIdx.x * 8 + wid;
  if (row >= (size_t)BB * TT) return;
  u16* p = P + row * SS;

  float v[SS / 32];
  float m = -INFINITY;
#pragma unroll
  for (int i = 0; i < SS / 32; ++i) {
    float x = bf2f(p[lane + 32 * i]);
    if (x == 0.0f) x = -INFINITY;   // reference's zero-mask hack
    v[i] = x;
    m = fmaxf(m, x);
  }
#pragma unroll
  for (int off = 16; off; off >>= 1) m = fmaxf(m, __shfl_xor(m, off, 32));
  float s = 0.f;
#pragma unroll
  for (int i = 0; i < SS / 32; ++i) { float e = __expf(v[i] - m); v[i] = e; s += e; }
#pragma unroll
  for (int off = 16; off; off >>= 1) s += __shfl_xor(s, off, 32);
  float inv = 1.0f / s;
#pragma unroll
  for (int i = 0; i < SS / 32; ++i) p[lane + 32 * i] = f2bf(v[i] * inv);
}

// ---------------- GEMM 2: mix = P . C  (bf16 out) --------------------------
__global__ __launch_bounds__(256) void k_gemm_av(const u16* __restrict__ sc,
                                                 const u16* __restrict__ cT,
                                                 u16* __restrict__ mixh) {
  __shared__ u16 As[2][BM * LDK];
  __shared__ u16 Bs[2][BN * LDK];
  const int NBm = TT / BM, NBn = DD / BN;   // 16, 4
  int bid = blockIdx.x;
  int b = bid / (NBm * NBn), rem = bid % (NBm * NBn);
  int mb = rem / NBn, nb = rem % NBn;

  f32x8 acc[4][4];
  zero_acc(acc);
  block_gemm(sc + (size_t)b * TT * SS + (size_t)(mb * BM) * SS, SS,
             cT + (size_t)b * DD * SS + (size_t)(nb * BN) * SS, SS,
             SS, As, Bs, acc);

  const int lane = threadIdx.x & 31, wid = threadIdx.x >> 5;
  const int wm = wid >> 2, wn = wid & 3;
  const int m0 = mb * BM + wm * 64, n0 = nb * BN + wn * 64;
  u16* C = mixh + (size_t)b * TT * DD;
  const int cn = lane & 15, hi = lane >> 4;
#pragma unroll
  for (int i = 0; i < 4; ++i)
#pragma unroll
    for (int j = 0; j < 4; ++j)
#pragma unroll
      for (int rr = 0; rr < 8; ++rr)
        C[(size_t)(m0 + i * 16 + rr + 8 * hi) * DD + (n0 + j * 16 + cn)] =
            f2bf(acc[i][j][rr]);
}

// ---------------- GEMM 3: out = tanh([mix, q] . W^T + b), fp32 out ---------
__global__ __launch_bounds__(256) void k_gemm_final(const u16* __restrict__ mixh,
                                                    const u16* __restrict__ qh,
                                                    const u16* __restrict__ Wh,
                                                    const float* __restrict__ bias,
                                                    float* __restrict__ out) {
  __shared__ u16 As[2][BM * LDK];
  __shared__ u16 Bs[2][BN * LDK];
  const int NBm = TT / BM, NBn = DD / BN;   // 16, 4
  int bid = blockIdx.x;
  int b = bid / (NBm * NBn), rem = bid % (NBm * NBn);
  int mb = rem / NBn, nb = rem % NBn;

  f32x8 acc[4][4];
  zero_acc(acc);
  // k in [0,DD): combined = mix, weight cols W[:, 0:DD]
  block_gemm(mixh + (size_t)b * TT * DD + (size_t)(mb * BM) * DD, DD,
             Wh + (size_t)(nb * BN) * (2 * DD), 2 * DD,
             DD, As, Bs, acc);
  // k in [DD,2DD): combined = q, weight cols W[:, DD:2DD]
  block_gemm(qh + (size_t)b * TT * DD + (size_t)(mb * BM) * DD, DD,
             Wh + (size_t)(nb * BN) * (2 * DD) + DD, 2 * DD,
             DD, As, Bs, acc);

  const int lane = threadIdx.x & 31, wid = threadIdx.x >> 5;
  const int wm = wid >> 2, wn = wid & 3;
  const int m0 = mb * BM + wm * 64, n0 = nb * BN + wn * 64;
  float* O = out + (size_t)b * TT * DD;
  const int cn = lane & 15, hi = lane >> 4;
#pragma unroll
  for (int j = 0; j < 4; ++j) {
    float bj = bias[n0 + j * 16 + cn];
#pragma unroll
    for (int i = 0; i < 4; ++i)
#pragma unroll
      for (int rr = 0; rr < 8; ++rr)
        O[(size_t)(m0 + i * 16 + rr + 8 * hi) * DD + (n0 + j * 16 + cn)] =
            tanhf(acc[i][j][rr] + bj);
  }
}

// ---------------- host launcher --------------------------------------------
extern "C" void kernel_launch(void* const* d_in, const int* in_sizes, int n_in,
                              void* d_out, int out_size, void* d_ws, size_t ws_size,
                              hipStream_t stream) {
  const float* output  = (const float*)d_in[0]; // [T,B,D]
  const float* context = (const float*)d_in[1]; // [S,B,D]
  const float* W       = (const float*)d_in[2]; // [D,2D]
  const float* bias    = (const float*)d_in[3]; // [D]
  float* out = (float*)d_out;                   // [B,T,D]

  char* ws = (char*)d_ws;
  const size_t sz_btd = (size_t)BB * TT * DD * 2;   // 64 MiB
  u16* qh    = (u16*)(ws);
  u16* ch    = (u16*)(ws + sz_btd);
  u16* cT    = (u16*)(ws + 2 * sz_btd);
  u16* sc    = (u16*)(ws + 3 * sz_btd);             // [B,T,S] bf16, 128 MiB
  u16* mixh  = (u16*)(ws + 3 * sz_btd + (size_t)BB * TT * SS * 2);
  u16* Wh    = (u16*)(ws + 4 * sz_btd + (size_t)BB * TT * SS * 2);

  // 1) precision/layout prep
  {
    size_t n = (size_t)TT * BB * DD / 8;
    k_cvt_swap<<<(unsigned)((n + 255) / 256), 256, 0, stream>>>(output, qh);
    k_cvt_swap<<<(unsigned)((n + 255) / 256), 256, 0, stream>>>(context, ch);
  }
  {
    size_t n = (size_t)BB * DD * SS / 8;
    k_cvt_T<<<(unsigned)((n + 255) / 256), 256, 0, stream>>>(context, cT);
  }
  {
    size_t n = (size_t)DD * 2 * DD / 8;
    k_cvt_w<<<(unsigned)((n + 255) / 256), 256, 0, stream>>>(W, Wh);
  }

  // 2) scores = Q C^T : blocks = B * (T/BM) * (S/BN)
  k_gemm_qk<<<BB * (TT / BM) * (SS / BN), 256, 0, stream>>>(qh, ch, sc);

  // 3) masked softmax rows
  k_softmax<<<(BB * TT) / 8, 256, 0, stream>>>(sc);

  // 4) mix = P C
  k_gemm_av<<<BB * (TT / BM) * (DD / BN), 256, 0, stream>>>(sc, cT, mixh);

  // 5) out = tanh([mix,q] W^T + b)
  k_gemm_final<<<BB * (TT / BM) * (DD / BN), 256, 0, stream>>>(mixh, qh, Wh, bias, out);
}
